// Encoder_57492432224815
// MI455X (gfx1250) — matrix operand order repaired and checked by reading, not compiled, not run on hardware
//
#include <hip/hip_runtime.h>
#include <hip/hip_bf16.h>

typedef __attribute__((ext_vector_type(16))) __bf16 v16bf;
typedef __attribute__((ext_vector_type(8)))  float  v8f;
typedef unsigned u32x4 __attribute__((ext_vector_type(4)));
typedef int      i32x8 __attribute__((ext_vector_type(8)));
typedef int      i32x4 __attribute__((ext_vector_type(4)));

// ---------- helpers ----------
__device__ __forceinline__ unsigned short f2bf(float f) {
  unsigned u = __builtin_bit_cast(unsigned, f);
  u += 0x7FFFu + ((u >> 16) & 1u);            // round-to-nearest-even
  return (unsigned short)(u >> 16);
}

__device__ __forceinline__ v8f wmma_bf16(v16bf a, v16bf b, v8f c) {
  return __builtin_amdgcn_wmma_f32_16x16x32_bf16(false, a, false, b, (short)0, c, false, false);
}

// A fragment: 16x32 tile, lane = row (mod 16), half selects K {0..7,16..23} vs {8..15,24..31}
__device__ __forceinline__ v16bf load_frag_a(const unsigned short* base, int ld, int koff, int lane) {
  int m = lane & 15, half = lane >> 4;
  const unsigned short* r = base + m * ld + koff + half * 8;
  union { v16bf v; unsigned u[8]; } f;
#pragma unroll
  for (int e = 0; e < 8; ++e)
    f.u[e] = *(const unsigned*)(r + ((e & 4) ? 16 : 0) + ((e & 3) << 1));
  return f.v;
}

// B fragment from N-major LDS tile (Bs[n][k]): lane = col, half selects K {0..7,16..23} vs {8..15,24..31}
__device__ __forceinline__ v16bf load_frag_b(const unsigned short* base, int ld, int koff, int lane) {
  int n = lane & 15, half = lane >> 4;
  const unsigned short* r = base + n * ld + koff + half * 8;
  union { v16bf v; unsigned u[8]; } f;
#pragma unroll
  for (int e = 0; e < 8; ++e)
    f.u[e] = *(const unsigned*)(r + ((e & 4) ? 16 : 0) + ((e & 3) << 1));
  return f.v;
}

// ---------- Tensor Data Mover: 2-D bf16 tile -> LDS with row padding ----------
// D# per CDNA5 ISA. pad codes: interval 2^(c+1) dwords, amount c+1 dwords.
// Toolchain uses the 6-arg builtin form: (g0, g1, g2, g3, g4, cpol).
__device__ __forceinline__ void tdm_load_2d_bf16(
    void* lds_dst, const unsigned short* gsrc,
    unsigned tile_k, unsigned tile_rows,
    unsigned tens_k, unsigned tens_rows, unsigned row_stride_elems,
    unsigned pad_interval_code, unsigned pad_amount_code) {
  unsigned lds_off = (unsigned)(size_t)lds_dst;           // flat LDS addr low bits = LDS offset
  unsigned long long ga = (unsigned long long)(size_t)gsrc;
  union { u32x4 v; unsigned w[4]; } g0;
  g0.w[0] = 1u;                                           // count=1, user descriptor
  g0.w[1] = lds_off;                                      // lds_addr
  g0.w[2] = (unsigned)ga;                                 // global_addr[31:0]
  g0.w[3] = (unsigned)(ga >> 32) | (2u << 30);            // global_addr[56:32] | type=2 (image)
  union { i32x8 v; unsigned w[8]; } g1;
  g1.w[0] = (1u << 16)                                    // data_size = 2 bytes
          | (1u << 20)                                    // pad_enable
          | (pad_interval_code << 22) | (pad_amount_code << 25);
  g1.w[1] = (tens_k & 0xFFFFu) << 16;                     // tensor_dim0[15:0]
  g1.w[2] = (tens_k >> 16) | ((tens_rows & 0xFFFFu) << 16);
  g1.w[3] = (tens_rows >> 16) | (tile_k << 16);           // tensor_dim1 hi | tile_dim0
  g1.w[4] = tile_rows & 0xFFFFu;                          // tile_dim1 (tile_dim2 = 0)
  g1.w[5] = row_stride_elems;                             // tensor_dim0_stride[31:0]
  g1.w[6] = 0u; g1.w[7] = 0u;
  i32x4 z4 = {0, 0, 0, 0};
  i32x8 z8 = {0, 0, 0, 0, 0, 0, 0, 0};
  __builtin_amdgcn_tensor_load_to_lds(g0.v, g1.v, z4, z4, z8, 0);
}

// ---------- embedding + positional encoding (fp32 residual + bf16 stream) ----------
__global__ __launch_bounds__(128) void embed_kernel(
    const int* __restrict__ src, const float* __restrict__ emb,
    const float* __restrict__ pe, float* __restrict__ x, unsigned short* __restrict__ xb) {
  int bs = blockIdx.x;
  int s = bs & 1023;
  int tok = src[bs];
  const float* er = emb + (size_t)tok * 512;
  const float* pr = pe + (size_t)s * 512;
  float* xr = x + (size_t)bs * 512;
  unsigned short* xbr = xb + (size_t)bs * 512;
#pragma unroll
  for (int i = 0; i < 4; ++i) {
    int c = threadIdx.x + 128 * i;
    float v = er[c] + pr[c];
    xr[c] = v;
    xbr[c] = f2bf(v);
  }
}

// ---------- fp32 [K,N] -> bf16 [N,K] transposed weight conversion (tiled) ----------
__global__ __launch_bounds__(256) void cvt_t_kernel(const float* __restrict__ w,
                                                    unsigned short* __restrict__ o,
                                                    int Kdim, int Ndim) {
  __shared__ unsigned short t[32][33];
  int n0 = blockIdx.x * 32, k0 = blockIdx.y * 32;
  int lx = threadIdx.x & 31, ly = threadIdx.x >> 5;
#pragma unroll
  for (int i = 0; i < 4; ++i) {
    int kk = ly + i * 8;
    t[kk][lx] = f2bf(w[(size_t)(k0 + kk) * Ndim + n0 + lx]);
  }
  __syncthreads();
#pragma unroll
  for (int i = 0; i < 4; ++i) {
    int nn = ly + i * 8;
    o[(size_t)(n0 + nn) * Kdim + k0 + lx] = t[lx][nn];
  }
}

// ---------- WMMA GEMM: C[M,N] = act(A[M,K]bf16 * Wt[N,K]bf16^T + bias) ----------
// 256 threads = 8 waves; block tile 128x64; K-step 32; TDM double-buffered LDS.
#define LDA 40
#define LDB 40
template <int RELU, int BF16OUT>
__global__ __launch_bounds__(256) void gemm_kernel(
    const unsigned short* __restrict__ A, const unsigned short* __restrict__ Wt,
    const float* __restrict__ bias, void* __restrict__ Cout,
    int M, int N, int K) {
  __shared__ __align__(16) unsigned short As[2][128][LDA];
  __shared__ __align__(16) unsigned short Bs[2][64][LDB];
  int tid = threadIdx.x, lane = tid & 31, w = tid >> 5;
  int wm = (w & 3) * 32, wn = (w >> 2) * 32;
  int m0 = blockIdx.y * 128, n0 = blockIdx.x * 64;
  v8f acc[2][2] = {};
  const int nk = K >> 5;
  if (w == 0) {
    tdm_load_2d_bf16(&As[0][0][0], A + (size_t)m0 * K, 32, 128, K, M, K, 3, 3);
    tdm_load_2d_bf16(&Bs[0][0][0], Wt + (size_t)n0 * K, 32, 64, K, N, K, 3, 3);
    __builtin_amdgcn_s_wait_tensorcnt(0);
  }
  __syncthreads();
  for (int t = 0; t < nk; ++t) {
    int cur = t & 1;
    if (w == 0 && t + 1 < nk) {                 // prefetch next K-slab while computing
      int k0 = (t + 1) << 5;
      tdm_load_2d_bf16(&As[cur ^ 1][0][0], A + (size_t)m0 * K + k0, 32, 128, K, M, K, 3, 3);
      tdm_load_2d_bf16(&Bs[cur ^ 1][0][0], Wt + (size_t)n0 * K + k0, 32, 64, K, N, K, 3, 3);
    }
    v16bf a0 = load_frag_a(&As[cur][wm][0],      LDA, 0, lane);
    v16bf a1 = load_frag_a(&As[cur][wm + 16][0], LDA, 0, lane);
    v16bf b0 = load_frag_b(&Bs[cur][wn][0],      LDB, 0, lane);
    v16bf b1 = load_frag_b(&Bs[cur][wn + 16][0], LDB, 0, lane);
    acc[0][0] = wmma_bf16(a0, b0, acc[0][0]);
    acc[0][1] = wmma_bf16(a0, b1, acc[0][1]);
    acc[1][0] = wmma_bf16(a1, b0, acc[1][0]);
    acc[1][1] = wmma_bf16(a1, b1, acc[1][1]);
    if (w == 0) __builtin_amdgcn_s_wait_tensorcnt(0);
    __syncthreads();
  }
  int half = lane >> 4, n15 = lane & 15;
#pragma unroll
  for (int i = 0; i < 2; ++i)
#pragma unroll
    for (int j = 0; j < 2; ++j) {
      int col = n0 + wn + j * 16 + n15;
      float bv = bias[col];
#pragma unroll
      for (int r = 0; r < 8; ++r) {
        int row = m0 + wm + i * 16 + r + half * 8;
        float v = acc[i][j][r] + bv;
        if (RELU) v = fmaxf(v, 0.f);
        if (BF16OUT) ((unsigned short*)Cout)[(size_t)row * N + col] = f2bf(v);
        else         ((float*)Cout)[(size_t)row * N + col] = v;
      }
    }
}

// ---------- fused flash-style attention (bf16 in/out, fp32 softmax state) ----------
// grid = B*H*(S/128); 256 threads; each wave owns 16 q-rows, full Dh=64.
__global__ __launch_bounds__(256) void attn_kernel(
    const unsigned short* __restrict__ Q, const unsigned short* __restrict__ Kt,
    const unsigned short* __restrict__ Vt, const int* __restrict__ mask,
    unsigned short* __restrict__ O) {
  __shared__ __align__(16) unsigned short Qs[128][72];
  __shared__ __align__(16) unsigned short Ks[64][72];
  __shared__ __align__(16) unsigned short Vs[64][72];   // stored [d][key]
  __shared__ __align__(16) unsigned short Ps[8][16][72];
  int tid = threadIdx.x, lane = tid & 31, w = tid >> 5;
  int half = lane >> 4, n15 = lane & 15;
  int bid = blockIdx.x;
  int qt = bid & 7, h = (bid >> 3) & 7, b = bid >> 6;
  int q0 = qt * 128;
  const unsigned short* Qg = Q + ((size_t)(b * 1024 + q0)) * 512 + h * 64;
  const unsigned short* Kg = Kt + ((size_t)b * 1024) * 512 + h * 64;
  const unsigned short* Vg = Vt + ((size_t)b * 1024) * 512 + h * 64;
  const int* mk = mask + b * 1024;

  if (w == 0) {                                   // TDM: Q tile 128x64, stride D, pad 32+4 dwords
    tdm_load_2d_bf16(&Qs[0][0], Qg, 64, 128, 64, 128, 512, 4, 3);
    __builtin_amdgcn_s_wait_tensorcnt(0);
  }
  __syncthreads();
  v16bf qf[2];
  qf[0] = load_frag_a(&Qs[w * 16][0], 72, 0, lane);
  qf[1] = load_frag_a(&Qs[w * 16][0], 72, 32, lane);

  v8f oacc[4] = {};
  float mrow[8], lrow[8];
#pragma unroll
  for (int r = 0; r < 8; ++r) { mrow[r] = -1e30f; lrow[r] = 0.f; }
  const float scale = 0.125f;                     // 1/sqrt(64)

  for (int kt = 0; kt < 16; ++kt) {
    int k0 = kt * 64;
    __syncthreads();                              // previous tile fully consumed
    if (w == 0)                                   // TDM: K tile 64x64
      tdm_load_2d_bf16(&Ks[0][0], Kg + (size_t)k0 * 512, 64, 64, 64, 64, 512, 4, 3);
#pragma unroll
    for (int i = 0; i < 16; ++i) {                // manual transposed V stage (bf16 copy)
      int li = tid + 256 * i;
      int r = li >> 6, c = li & 63;
      Vs[c][r] = Vg[(size_t)(k0 + r) * 512 + c];
    }
    if (w == 0) __builtin_amdgcn_s_wait_tensorcnt(0);
    __syncthreads();

    v8f sacc[4] = {};
#pragma unroll
    for (int ks = 0; ks < 2; ++ks)
#pragma unroll
      for (int j = 0; j < 4; ++j)
        sacc[j] = wmma_bf16(qf[ks], load_frag_b(&Ks[j * 16][0], 72, ks * 32, lane), sacc[j]);

    float mv[4];
#pragma unroll
    for (int j = 0; j < 4; ++j) mv[j] = (mk[k0 + j * 16 + n15] == 0) ? -1e30f : 0.f;

#pragma unroll
    for (int r = 0; r < 8; ++r) {                 // online softmax per q-row
      float s[4], rowm = -1e30f;
#pragma unroll
      for (int j = 0; j < 4; ++j) { s[j] = sacc[j][r] * scale + mv[j]; rowm = fmaxf(rowm, s[j]); }
#pragma unroll
      for (int m = 8; m >= 1; m >>= 1) rowm = fmaxf(rowm, __shfl_xor(rowm, m, 32));
      float mnew = fmaxf(mrow[r], rowm);
      float corr = __expf(mrow[r] - mnew);
      float rs = 0.f;
#pragma unroll
      for (int j = 0; j < 4; ++j) {
        float pv = __expf(s[j] - mnew);
        rs += pv;
        Ps[w][r + half * 8][j * 16 + n15] = f2bf(pv);
      }
#pragma unroll
      for (int m = 8; m >= 1; m >>= 1) rs += __shfl_xor(rs, m, 32);
      lrow[r] = lrow[r] * corr + rs;
      mrow[r] = mnew;
#pragma unroll
      for (int j = 0; j < 4; ++j) oacc[j][r] *= corr;
    }
    asm volatile("s_wait_dscnt 0" ::: "memory");  // P store -> P fragment reload (same wave)
#pragma unroll
    for (int ks = 0; ks < 2; ++ks) {
      v16bf pa = load_frag_a(&Ps[w][0][0], 72, ks * 32, lane);
#pragma unroll
      for (int j = 0; j < 4; ++j)
        oacc[j] = wmma_bf16(pa, load_frag_b(&Vs[j * 16][0], 72, ks * 32, lane), oacc[j]);
    }
  }

  unsigned short* Og = O + ((size_t)(b * 1024 + q0)) * 512 + h * 64;
#pragma unroll
  for (int j = 0; j < 4; ++j)
#pragma unroll
    for (int r = 0; r < 8; ++r) {
      int row = w * 16 + r + half * 8;
      Og[(size_t)row * 512 + j * 16 + n15] = f2bf(oacc[j][r] / lrow[r]);
    }
}

// ---------- residual add + LayerNorm (wave per row), fp32 + bf16 outputs ----------
__global__ __launch_bounds__(256) void add_ln_kernel(
    const float* __restrict__ a, const float* __restrict__ c,
    const float* __restrict__ g, const float* __restrict__ be,
    float* __restrict__ out, unsigned short* __restrict__ outb) {
  int lane = threadIdx.x & 31, w = threadIdx.x >> 5;
  int row = blockIdx.x * 8 + w;
  const float* ar = a + (size_t)row * 512;
  const float* cr = c + (size_t)row * 512;
  float t[16], s = 0.f, ss = 0.f;
#pragma unroll
  for (int i = 0; i < 16; ++i) {
    int idx = lane + 32 * i;
    float v = ar[idx] + cr[idx];
    t[i] = v; s += v; ss += v * v;
  }
#pragma unroll
  for (int m = 16; m >= 1; m >>= 1) { s += __shfl_xor(s, m, 32); ss += __shfl_xor(ss, m, 32); }
  float mu = s * (1.f / 512.f);
  float var = ss * (1.f / 512.f) - mu * mu;
  float rs = rsqrtf(var + 1e-5f);
  float* orow = out + (size_t)row * 512;
  unsigned short* obrow = outb + (size_t)row * 512;
#pragma unroll
  for (int i = 0; i < 16; ++i) {
    int idx = lane + 32 * i;
    float v = (t[i] - mu) * rs * g[idx] + be[idx];
    orow[idx] = v;
    obrow[idx] = f2bf(v);
  }
}

// ---------- orchestration ----------
extern "C" void kernel_launch(void* const* d_in, const int* in_sizes, int n_in,
                              void* d_out, int out_size, void* d_ws, size_t ws_size,
                              hipStream_t stream) {
  (void)in_sizes; (void)n_in; (void)out_size; (void)ws_size;
  const int*   src  = (const int*)d_in[0];
  const int*   mask = (const int*)d_in[1];
  const float* emb  = (const float*)d_in[2];
  const float* pe   = (const float*)d_in[3];
  const float* wq = (const float*)d_in[4];  const float* bq  = (const float*)d_in[5];
  const float* wk = (const float*)d_in[6];  const float* bk  = (const float*)d_in[7];
  const float* wv = (const float*)d_in[8];  const float* bv  = (const float*)d_in[9];
  const float* wo = (const float*)d_in[10]; const float* bo  = (const float*)d_in[11];
  const float* w1 = (const float*)d_in[12]; const float* b1  = (const float*)d_in[13];
  const float* w2 = (const float*)d_in[14]; const float* b2  = (const float*)d_in[15];
  const float* g1 = (const float*)d_in[16]; const float* be1 = (const float*)d_in[17];
  const float* g2 = (const float*)d_in[18]; const float* be2 = (const float*)d_in[19];

  const size_t SD = (size_t)8 * 1024 * 512;
  const size_t SF = (size_t)8 * 1024 * 2048;
  const size_t DD = (size_t)512 * 512;
  const size_t DF = (size_t)512 * 2048;
  char* p = (char*)d_ws;
  float* X  = (float*)p; p += SD * 4;
  float* X1 = (float*)p; p += SD * 4;
  float* P1 = (float*)p; p += SD * 4;
  unsigned short* Xb   = (unsigned short*)p; p += SD * 2;
  unsigned short* X1b  = (unsigned short*)p; p += SD * 2;
  unsigned short* Qb   = (unsigned short*)p; p += SD * 2;
  unsigned short* Kb   = (unsigned short*)p; p += SD * 2;
  unsigned short* Vb   = (unsigned short*)p; p += SD * 2;
  unsigned short* ATTb = (unsigned short*)p; p += SD * 2;
  unsigned short* FF16 = (unsigned short*)p; p += SF * 2;
  unsigned short* wqt = (unsigned short*)p; p += DD * 2;
  unsigned short* wkt = (unsigned short*)p; p += DD * 2;
  unsigned short* wvt = (unsigned short*)p; p += DD * 2;
  unsigned short* wot = (unsigned short*)p; p += DD * 2;
  unsigned short* w1t = (unsigned short*)p; p += DF * 2;
  unsigned short* w2t = (unsigned short*)p; p += DF * 2;

  embed_kernel<<<8192, 128, 0, stream>>>(src, emb, pe, X, Xb);

  dim3 g512(8, 64), g2048(32, 64);
  for (int l = 0; l < 6; ++l) {
    cvt_t_kernel<<<dim3(16, 16), 256, 0, stream>>>(wq + (size_t)l * DD, wqt, 512, 512);
    cvt_t_kernel<<<dim3(16, 16), 256, 0, stream>>>(wk + (size_t)l * DD, wkt, 512, 512);
    cvt_t_kernel<<<dim3(16, 16), 256, 0, stream>>>(wv + (size_t)l * DD, wvt, 512, 512);
    cvt_t_kernel<<<dim3(16, 16), 256, 0, stream>>>(wo + (size_t)l * DD, wot, 512, 512);
    cvt_t_kernel<<<dim3(64, 16), 256, 0, stream>>>(w1 + (size_t)l * DF, w1t, 512, 2048);
    cvt_t_kernel<<<dim3(16, 64), 256, 0, stream>>>(w2 + (size_t)l * DF, w2t, 2048, 512);

    gemm_kernel<0, 1><<<g512, 256, 0, stream>>>(Xb, wqt, bq + l * 512, Qb, 8192, 512, 512);
    gemm_kernel<0, 1><<<g512, 256, 0, stream>>>(Xb, wkt, bk + l * 512, Kb, 8192, 512, 512);
    gemm_kernel<0, 1><<<g512, 256, 0, stream>>>(Xb, wvt, bv + l * 512, Vb, 8192, 512, 512);

    attn_kernel<<<512, 256, 0, stream>>>(Qb, Kb, Vb, mask, ATTb);

    gemm_kernel<0, 0><<<g512, 256, 0, stream>>>(ATTb, wot, bo + l * 512, P1, 8192, 512, 512);
    add_ln_kernel<<<1024, 256, 0, stream>>>(X, P1, g1 + l * 512, be1 + l * 512, X1, X1b);

    gemm_kernel<1, 1><<<g2048, 256, 0, stream>>>(X1b, w1t, b1 + (size_t)l * 2048, FF16, 8192, 2048, 512);
    gemm_kernel<0, 0><<<g512, 256, 0, stream>>>(FF16, w2t, b2 + l * 512, P1, 8192, 512, 2048);

    float* dst = (l == 5) ? (float*)d_out : X;
    add_ln_kernel<<<1024, 256, 0, stream>>>(X1, P1, g2 + l * 512, be2 + l * 512, dst, Xb);
  }
}
